// BitNetACMoEGAModel_43576738185545
// MI455X (gfx1250) — compile-verified, matrix operand order checked by reading, not run on hardware
//
#include <hip/hip_runtime.h>
#include <hip/hip_bf16.h>

#define BROWS 131072

typedef __attribute__((ext_vector_type(16))) _Float16 v16h;
typedef __attribute__((ext_vector_type(8)))  float    v8f;
typedef unsigned int u32;
typedef __attribute__((ext_vector_type(4))) u32 u32x4;
typedef __attribute__((ext_vector_type(8))) int i32x8;
typedef __attribute__((ext_vector_type(4))) int i32x4;

__device__ __forceinline__ float sigm_f(float x) { return 1.0f / (1.0f + expf(-x)); }
__device__ __forceinline__ float gelu_f(float x) { return 0.5f * x * (1.0f + erff(x * 0.70710678118654752f)); }

// ---------------------------------------------------------------------------
// Weight prep: BitNet ternary quantization (single block per matrix)
// ---------------------------------------------------------------------------
__global__ void quant_f16_kernel(const float* __restrict__ W, _Float16* __restrict__ Q, int n) {
    __shared__ float red[256];
    float s = 0.0f;
    for (int i = threadIdx.x; i < n; i += 256) s += fabsf(W[i]);
    red[threadIdx.x] = s;
    __syncthreads();
    for (int st = 128; st > 0; st >>= 1) {
        if ((int)threadIdx.x < st) red[threadIdx.x] += red[threadIdx.x + st];
        __syncthreads();
    }
    float sc = fmaxf(red[0] / (float)n, 1e-5f);
    for (int i = threadIdx.x; i < n; i += 256) {
        float q = fminf(fmaxf(rintf(W[i] / sc), -1.0f), 1.0f) * sc;
        Q[i] = (_Float16)q;
    }
}

__global__ void quant_f32_kernel(const float* __restrict__ W, float* __restrict__ Q, int n) {
    __shared__ float red[256];
    float s = 0.0f;
    for (int i = threadIdx.x; i < n; i += 256) s += fabsf(W[i]);
    red[threadIdx.x] = s;
    __syncthreads();
    for (int st = 128; st > 0; st >>= 1) {
        if ((int)threadIdx.x < st) red[threadIdx.x] += red[threadIdx.x + st];
        __syncthreads();
    }
    float sc = fmaxf(red[0] / (float)n, 1e-5f);
    for (int i = threadIdx.x; i < n; i += 256) {
        Q[i] = fminf(fmaxf(rintf(W[i] / sc), -1.0f), 1.0f) * sc;
    }
}

// Plain f32 -> f16 convert (m1W / m2W are NOT quantized in the reference)
__global__ void conv_f16_kernel(const float* __restrict__ W, _Float16* __restrict__ Q, int n) {
    int i = blockIdx.x * blockDim.x + threadIdx.x;
    if (i < n) Q[i] = (_Float16)W[i];
}

// ---------------------------------------------------------------------------
// Encoder: one thread per row -> comb[B][256] in f16
// ---------------------------------------------------------------------------
struct EncP {
    const int *low8, *high8, *align_, *small_, *delta, *hamm;
    const float *continuous, *bitfield, *sketch;
    const int *ph, *off, *cl, *aa, *str, *rd, *lc, *ent;
    const float *aflags;
    const int *et, *fc, *sc, *of, *tt, *rc;
    const float *pte;
    const int *vma, *prot;
    const float *zsum;
    const float *E_low8, *E_high8, *E_align, *E_small, *E_delta, *E_hamm;
    const float *bc, *bb, *bs, *gb_b, *tb_b, *lng_b, *lnb_b;
    const float *E_ph, *E_off, *E_cl, *E_aa, *E_str, *E_rd, *E_lc, *E_ent;
    const float *bf, *gb_a, *tb_a, *lng_a, *lnb_a;
    const float *E_et, *E_fc, *E_sc, *E_of, *E_tt, *E_rc, *b_e, *lng_e, *lnb_e;
    const float *E_vma, *E_prot, *bp, *gb_m, *tb_m, *lng_m, *lnb_m;
    const float *Wc, *Wb, *Ws, *gW_b, *tW_b, *Wf, *gW_a, *tW_a, *W_e, *Wp, *gW_m, *tW_m;
    _Float16* comb;
};

template <int IN, int OUT>
__device__ void gfuse_ln(const float* c, const float* gW, const float* gb,
                         const float* tW, const float* tb,
                         const float* lg, const float* lb, float* z) {
    float m = 0.0f;
    for (int o = 0; o < OUT; ++o) {
        float g = gb[o], t = tb[o];
        for (int i = 0; i < IN; ++i) {
            g += gW[o * IN + i] * c[i];
            t += tW[o * IN + i] * c[i];
        }
        float v = sigm_f(g) * t;
        z[o] = v;
        m += v;
    }
    m /= (float)OUT;
    float var = 0.0f;
    for (int o = 0; o < OUT; ++o) { float d = z[o] - m; var += d * d; }
    var /= (float)OUT;
    float rstd = rsqrtf(var + 1e-5f);
    for (int o = 0; o < OUT; ++o) z[o] = (z[o] - m) * rstd * lg[o] + lb[o];
}

template <int IN, int OUT>
__device__ void tlin_ln(const float* c, const float* W, const float* b,
                        const float* lg, const float* lb, float* z) {
    float m = 0.0f;
    for (int o = 0; o < OUT; ++o) {
        float t = b[o];
        for (int i = 0; i < IN; ++i) t += W[o * IN + i] * c[i];
        z[o] = t;
        m += t;
    }
    m /= (float)OUT;
    float var = 0.0f;
    for (int o = 0; o < OUT; ++o) { float d = z[o] - m; var += d * d; }
    var /= (float)OUT;
    float rstd = rsqrtf(var + 1e-5f);
    for (int o = 0; o < OUT; ++o) z[o] = (z[o] - m) * rstd * lg[o] + lb[o];
}

__global__ void encoder_kernel(EncP p) {
    int r = blockIdx.x * blockDim.x + threadIdx.x;
    if (r >= BROWS) return;
    _Float16* combRow = p.comb + (size_t)r * 256;

    // ---- byte plane (cb: 62) ----
    {
        float cb[62];
        int id;
        id = p.low8[r];   for (int j = 0; j < 8; ++j) cb[j]      = p.E_low8[id * 8 + j];
        id = p.high8[r];  for (int j = 0; j < 8; ++j) cb[8 + j]  = p.E_high8[id * 8 + j];
        id = p.align_[r]; for (int j = 0; j < 4; ++j) cb[16 + j] = p.E_align[id * 4 + j];
        id = p.small_[r]; for (int j = 0; j < 4; ++j) cb[20 + j] = p.E_small[id * 4 + j];
        id = p.delta[r];  for (int j = 0; j < 6; ++j) cb[24 + j] = p.E_delta[id * 6 + j];
        id = p.hamm[r];   for (int j = 0; j < 4; ++j) cb[30 + j] = p.E_hamm[id * 4 + j];
        for (int o = 0; o < 8; ++o) {
            float a = p.bc[o];
            for (int i = 0; i < 6; ++i) a += p.Wc[o * 6 + i] * p.continuous[(size_t)r * 6 + i];
            cb[34 + o] = a;
        }
        for (int o = 0; o < 8; ++o) {
            float a = p.bb[o];
            for (int i = 0; i < 16; ++i) a += p.Wb[o * 16 + i] * p.bitfield[(size_t)r * 16 + i];
            cb[42 + o] = a;
        }
        for (int o = 0; o < 12; ++o) {
            float a = p.bs[o];
            for (int i = 0; i < 32; ++i) a += p.Ws[o * 32 + i] * p.sketch[(size_t)r * 32 + i];
            cb[50 + o] = a;
        }
        float zb[64];
        gfuse_ln<62, 64>(cb, p.gW_b, p.gb_b, p.tW_b, p.tb_b, p.lng_b, p.lnb_b, zb);
        for (int o = 0; o < 64; ++o) combRow[o] = (_Float16)zb[o];
    }

    // ---- address plane (ca: 44) ----
    {
        float ca[44];
        int id;
        id = p.ph[r];  for (int j = 0; j < 8; ++j) ca[j]      = p.E_ph[id * 8 + j];
        id = p.off[r]; for (int j = 0; j < 4; ++j) ca[8 + j]  = p.E_off[id * 4 + j];
        id = p.cl[r];  for (int j = 0; j < 4; ++j) ca[12 + j] = p.E_cl[id * 4 + j];
        id = p.aa[r];  for (int j = 0; j < 3; ++j) ca[16 + j] = p.E_aa[id * 3 + j];
        id = p.str[r]; for (int j = 0; j < 6; ++j) ca[19 + j] = p.E_str[id * 6 + j];
        id = p.rd[r];  for (int j = 0; j < 5; ++j) ca[25 + j] = p.E_rd[id * 5 + j];
        id = p.lc[r];  for (int j = 0; j < 6; ++j) ca[30 + j] = p.E_lc[id * 6 + j];
        id = p.ent[r]; for (int j = 0; j < 3; ++j) ca[36 + j] = p.E_ent[id * 3 + j];
        for (int o = 0; o < 5; ++o) {
            float a = p.bf[o];
            for (int i = 0; i < 5; ++i) a += p.Wf[o * 5 + i] * p.aflags[(size_t)r * 5 + i];
            ca[39 + o] = a;
        }
        float za[64];
        gfuse_ln<44, 64>(ca, p.gW_a, p.gb_a, p.tW_a, p.tb_a, p.lng_a, p.lnb_a, za);
        for (int o = 0; o < 64; ++o) combRow[64 + o] = (_Float16)za[o];
    }

    // ---- event plane (ce: 96) ----
    {
        float ce[96];
        int id;
        id = p.et[r]; for (int j = 0; j < 16; ++j) ce[j]      = p.E_et[id * 16 + j];
        id = p.fc[r]; for (int j = 0; j < 16; ++j) ce[16 + j] = p.E_fc[id * 16 + j];
        id = p.sc[r]; for (int j = 0; j < 16; ++j) ce[32 + j] = p.E_sc[id * 16 + j];
        id = p.of[r]; for (int j = 0; j < 16; ++j) ce[48 + j] = p.E_of[id * 16 + j];
        id = p.tt[r]; for (int j = 0; j < 16; ++j) ce[64 + j] = p.E_tt[id * 16 + j];
        id = p.rc[r]; for (int j = 0; j < 16; ++j) ce[80 + j] = p.E_rc[id * 16 + j];
        float ze[64];
        tlin_ln<96, 64>(ce, p.W_e, p.b_e, p.lng_e, p.lnb_e, ze);
        for (int o = 0; o < 64; ++o) combRow[128 + o] = (_Float16)ze[o];
    }

    // ---- map plane (cm: 15) ----
    {
        float cm[15];
        int id;
        id = p.vma[r];  for (int j = 0; j < 4; ++j) cm[j]     = p.E_vma[id * 4 + j];
        id = p.prot[r]; for (int j = 0; j < 3; ++j) cm[4 + j] = p.E_prot[id * 3 + j];
        for (int o = 0; o < 8; ++o) {
            float a = p.bp[o];
            for (int i = 0; i < 11; ++i) a += p.Wp[o * 11 + i] * p.pte[(size_t)r * 11 + i];
            cm[7 + o] = a;
        }
        float zm[32];
        gfuse_ln<15, 32>(cm, p.gW_m, p.gb_m, p.tW_m, p.tb_m, p.lng_m, p.lnb_m, zm);
        for (int o = 0; o < 32; ++o) combRow[192 + o] = (_Float16)zm[o];
    }

    // ---- z_sum passthrough ----
    for (int j = 0; j < 32; ++j) combRow[224 + j] = (_Float16)p.zsum[(size_t)r * 32 + j];
}

// ---------------------------------------------------------------------------
// Fused observation block: all GEMMs via v_wmma_f32_16x16x32_f16.
// Block = 256 threads = 8 wave32s, 32 rows per block.
// The 32x256 f16 activation tile is DMA'd into LDS by the Tensor Data Mover.
// ---------------------------------------------------------------------------
struct GemmP {
    const _Float16 *comb, *g1q, *g2q, *t1q, *t2q, *m1q, *m2q;
    const float *g1b, *g2b, *t1b, *t2b, *m1b, *m2b;
    const float *ln1g, *ln1b, *ln2g, *ln2b;
    float* out;
};

__global__ void fused_block_kernel(GemmP p) {
    extern __shared__ _Float16 smem[];
    _Float16* combS  = smem;                 // 32 x 256  (filled by TDM)
    _Float16* tmpS   = combS + 32 * 256;     // 32 x 512
    _Float16* fusedS = tmpS + 32 * 512;      // 32 x 256
    _Float16* outS   = fusedS + 32 * 256;    // 32 x 256

    const int tid  = threadIdx.x;
    const int lane = tid & 31;
    const int wv   = tid >> 5;
    const int ln16 = lane & 15;
    const int lhi  = lane >> 4;
    const int rowBase = blockIdx.x * 32;

    // Warm L2/WGP$ for the weight streams (global_prefetch_b8)
    __builtin_prefetch(p.g1q, 0, 1);
    __builtin_prefetch(p.t1q, 0, 1);
    __builtin_prefetch(p.m1q, 0, 1);

    // ---- Stage 0: TDM DMA of the comb activation tile into LDS ----
#if __has_builtin(__builtin_amdgcn_tensor_load_to_lds) && __has_builtin(__builtin_amdgcn_s_wait_tensorcnt)
    if (tid == 0) {
        unsigned long long gaddr =
            (unsigned long long)(__UINTPTR_TYPE__)(p.comb + (size_t)rowBase * 256);
        u32 ldsoff = __builtin_amdgcn_groupstaticsize();   // combS sits at dynamic-LDS base
        // D# group0: count=1 | lds_addr | global_addr | type=2
        u32x4 g0;
        g0.x = 0x1u;
        g0.y = ldsoff;
        g0.z = (u32)gaddr;
        g0.w = (u32)(gaddr >> 32) | 0x80000000u;           // type=2 in bits [127:126]
        // D# group1: data_size=2B; tensor_dim0=256; tensor_dim1=131072 (0x20000);
        //            tile_dim0=256; tile_dim1=32; tensor_dim0_stride=256
        i32x8 g1;
        g1[0] = 0x00010000;                 // [17:16] data_size = 1 -> 2 bytes
        g1[1] = (int)(256u << 16);          // tensor_dim0 low16 at bits [31:16] of word1
        g1[2] = 0;                          // tensor_dim0 hi16 = 0, tensor_dim1 low16 = 0
        g1[3] = (int)((256u << 16) | 2u);   // tensor_dim1 hi16 = 2 (131072); tile_dim0 = 256
        g1[4] = 32;                         // tile_dim1 = 32, tile_dim2 = 0
        g1[5] = 256;                        // tensor_dim0_stride = 256 (low 32)
        g1[6] = 0;
        g1[7] = 0;
        i32x4 gz4 = {0, 0, 0, 0};
        i32x8 gz8 = {0, 0, 0, 0, 0, 0, 0, 0};
        __builtin_amdgcn_tensor_load_to_lds(g0, g1, gz4, gz4, gz8, 0);
        __builtin_amdgcn_s_wait_tensorcnt(0);
    }
#else
    // Fallback: cooperative copy (32 f16 per thread)
    {
        const _Float16* src = p.comb + (size_t)rowBase * 256;
        for (int i = tid; i < 32 * 256; i += 256) combS[i] = src[i];
    }
#endif
    __syncthreads();

    // ---- Stage 1: ghid = gelu(comb@g1W^T+g1b), thid = gelu(comb@t1W^T+t1b) ----
    for (int t = wv; t < 64; t += 8) {
        const int path = t >> 5;              // 0 = gate hidden, 1 = trans hidden
        const int tt = t & 31, mt = tt >> 4, nt = tt & 15;
        const int row0 = mt * 16, col0 = nt * 16;
        const _Float16* Wq   = path ? p.t1q : p.g1q;
        const float*    bias = path ? p.t1b : p.g1b;
        const _Float16* Abase = combS + (row0 + ln16) * 256 + lhi * 16;
        const _Float16* Bbase = Wq + (size_t)(col0 + ln16) * 256 + lhi * 16;
        v8f c = {};
        for (int k = 0; k < 256; k += 32) {
            v16h a = *(const v16h*)(Abase + k);
            v16h b = *(const v16h*)(Bbase + k);
            c = __builtin_amdgcn_wmma_f32_16x16x32_f16(false, a, false, b, (short)0, c, false, false);
        }
        const int col = col0 + ln16;
        const float bi = bias[col];
        for (int rr = 0; rr < 8; ++rr) {
            int row = row0 + rr + lhi * 8;
            tmpS[row * 512 + path * 256 + col] = (_Float16)gelu_f(c[rr] + bi);
        }
    }
    __syncthreads();

    // ---- Stage 2a: trans = thid@t2W^T + t2b -> fusedS ----
    for (int t = wv; t < 32; t += 8) {
        const int mt = t >> 4, nt = t & 15;
        const int row0 = mt * 16, col0 = nt * 16;
        const _Float16* Abase = tmpS + (row0 + ln16) * 512 + 256 + lhi * 16;
        const _Float16* Bbase = p.t2q + (size_t)(col0 + ln16) * 256 + lhi * 16;
        v8f c = {};
        for (int k = 0; k < 256; k += 32) {
            v16h a = *(const v16h*)(Abase + k);
            v16h b = *(const v16h*)(Bbase + k);
            c = __builtin_amdgcn_wmma_f32_16x16x32_f16(false, a, false, b, (short)0, c, false, false);
        }
        const int col = col0 + ln16;
        const float bi = p.t2b[col];
        for (int rr = 0; rr < 8; ++rr) {
            int row = row0 + rr + lhi * 8;
            fusedS[row * 256 + col] = (_Float16)(c[rr] + bi);
        }
    }
    __syncthreads();

    // ---- Stage 2b: gate = sigmoid(1.2*(ghid@g2W^T+g2b)); fusedS *= gate ----
    for (int t = wv; t < 32; t += 8) {
        const int mt = t >> 4, nt = t & 15;
        const int row0 = mt * 16, col0 = nt * 16;
        const _Float16* Abase = tmpS + (row0 + ln16) * 512 + lhi * 16;
        const _Float16* Bbase = p.g2q + (size_t)(col0 + ln16) * 256 + lhi * 16;
        v8f c = {};
        for (int k = 0; k < 256; k += 32) {
            v16h a = *(const v16h*)(Abase + k);
            v16h b = *(const v16h*)(Bbase + k);
            c = __builtin_amdgcn_wmma_f32_16x16x32_f16(false, a, false, b, (short)0, c, false, false);
        }
        const int col = col0 + ln16;
        const float bi = p.g2b[col];
        for (int rr = 0; rr < 8; ++rr) {
            int row = row0 + rr + lhi * 8;
            float g = sigm_f(1.2f * (c[rr] + bi));
            int idx = row * 256 + col;
            fusedS[idx] = (_Float16)(g * (float)fusedS[idx]);
        }
    }
    __syncthreads();

    // ---- LN1 over fusedS rows (in place) ----
    if (tid < 32) {
        const int row = tid;
        float m = 0.0f;
        for (int cidx = 0; cidx < 256; ++cidx) m += (float)fusedS[row * 256 + cidx];
        m *= (1.0f / 256.0f);
        float var = 0.0f;
        for (int cidx = 0; cidx < 256; ++cidx) {
            float d = (float)fusedS[row * 256 + cidx] - m;
            var += d * d;
        }
        var *= (1.0f / 256.0f);
        float rstd = rsqrtf(var + 1e-5f);
        for (int cidx = 0; cidx < 256; ++cidx) {
            float v = ((float)fusedS[row * 256 + cidx] - m) * rstd * p.ln1g[cidx] + p.ln1b[cidx];
            fusedS[row * 256 + cidx] = (_Float16)v;
        }
    }
    __syncthreads();

    // ---- Stage 3: hid = gelu(fused@m1W^T + m1b), N=512 -> tmpS ----
    for (int t = wv; t < 64; t += 8) {
        const int mt = t >> 5, nt = t & 31;
        const int row0 = mt * 16, col0 = nt * 16;
        const _Float16* Abase = fusedS + (row0 + ln16) * 256 + lhi * 16;
        const _Float16* Bbase = p.m1q + (size_t)(col0 + ln16) * 256 + lhi * 16;
        v8f c = {};
        for (int k = 0; k < 256; k += 32) {
            v16h a = *(const v16h*)(Abase + k);
            v16h b = *(const v16h*)(Bbase + k);
            c = __builtin_amdgcn_wmma_f32_16x16x32_f16(false, a, false, b, (short)0, c, false, false);
        }
        const int col = col0 + ln16;
        const float bi = p.m1b[col];
        for (int rr = 0; rr < 8; ++rr) {
            int row = row0 + rr + lhi * 8;
            tmpS[row * 512 + col] = (_Float16)gelu_f(c[rr] + bi);
        }
    }
    __syncthreads();

    // ---- Stage 4: mlp = hid@m2W^T + m2b, K=512; residual add -> outS ----
    for (int t = wv; t < 32; t += 8) {
        const int mt = t >> 4, nt = t & 15;
        const int row0 = mt * 16, col0 = nt * 16;
        const _Float16* Abase = tmpS + (row0 + ln16) * 512 + lhi * 16;
        const _Float16* Bbase = p.m2q + (size_t)(col0 + ln16) * 512 + lhi * 16;
        v8f c = {};
        for (int k = 0; k < 512; k += 32) {
            v16h a = *(const v16h*)(Abase + k);
            v16h b = *(const v16h*)(Bbase + k);
            c = __builtin_amdgcn_wmma_f32_16x16x32_f16(false, a, false, b, (short)0, c, false, false);
        }
        const int col = col0 + ln16;
        const float bi = p.m2b[col];
        for (int rr = 0; rr < 8; ++rr) {
            int row = row0 + rr + lhi * 8;
            float res = c[rr] + bi + (float)fusedS[row * 256 + col];
            outS[row * 256 + col] = (_Float16)res;
        }
    }
    __syncthreads();

    // ---- LN2 over outS rows -> global f32 output ----
    if (tid < 32) {
        const int row = tid;
        float m = 0.0f;
        for (int cidx = 0; cidx < 256; ++cidx) m += (float)outS[row * 256 + cidx];
        m *= (1.0f / 256.0f);
        float var = 0.0f;
        for (int cidx = 0; cidx < 256; ++cidx) {
            float d = (float)outS[row * 256 + cidx] - m;
            var += d * d;
        }
        var *= (1.0f / 256.0f);
        float rstd = rsqrtf(var + 1e-5f);
        float* orow = p.out + (size_t)(rowBase + row) * 256;
        for (int cidx = 0; cidx < 256; ++cidx) {
            orow[cidx] = ((float)outS[row * 256 + cidx] - m) * rstd * p.ln2g[cidx] + p.ln2b[cidx];
        }
    }
}

// ---------------------------------------------------------------------------
// Host launcher
// ---------------------------------------------------------------------------
extern "C" void kernel_launch(void* const* d_in, const int* in_sizes, int n_in,
                              void* d_out, int out_size, void* d_ws, size_t ws_size,
                              hipStream_t stream) {
    (void)in_sizes; (void)n_in; (void)out_size; (void)ws_size;

    // ---- workspace carve-up ----
    char* ws = (char*)d_ws;
    size_t off = 0;
    _Float16* comb = (_Float16*)(ws + off); off += (size_t)BROWS * 256 * sizeof(_Float16);
    _Float16* g1q = (_Float16*)(ws + off); off += 256 * 256 * sizeof(_Float16);
    _Float16* g2q = (_Float16*)(ws + off); off += 256 * 256 * sizeof(_Float16);
    _Float16* t1q = (_Float16*)(ws + off); off += 256 * 256 * sizeof(_Float16);
    _Float16* t2q = (_Float16*)(ws + off); off += 256 * 256 * sizeof(_Float16);
    _Float16* m1q = (_Float16*)(ws + off); off += 512 * 256 * sizeof(_Float16);
    _Float16* m2q = (_Float16*)(ws + off); off += 256 * 512 * sizeof(_Float16);
    float* WcQ  = (float*)(ws + off); off += 8 * 6 * 4;
    float* WbQ  = (float*)(ws + off); off += 8 * 16 * 4;
    float* WsQ  = (float*)(ws + off); off += 12 * 32 * 4;
    float* gWbQ = (float*)(ws + off); off += 64 * 62 * 4;
    float* tWbQ = (float*)(ws + off); off += 64 * 62 * 4;
    float* WfQ  = (float*)(ws + off); off += 5 * 5 * 4;
    float* gWaQ = (float*)(ws + off); off += 64 * 44 * 4;
    float* tWaQ = (float*)(ws + off); off += 64 * 44 * 4;
    float* WeQ  = (float*)(ws + off); off += 64 * 96 * 4;
    float* WpQ  = (float*)(ws + off); off += 8 * 11 * 4;
    float* gWmQ = (float*)(ws + off); off += 32 * 15 * 4;
    float* tWmQ = (float*)(ws + off); off += 32 * 15 * 4;

    // data: 0..27 | byte: 28..45 | addr: 46..61 | evt: 62..71 | map: 72..81 | fused: 82..97
    #define IN_F(i) ((const float*)d_in[(i)])
    #define IN_I(i) ((const int*)d_in[(i)])

    // ---- weight prep ----
    quant_f32_kernel<<<1, 256, 0, stream>>>(IN_F(34), WcQ, 8 * 6);
    quant_f32_kernel<<<1, 256, 0, stream>>>(IN_F(36), WbQ, 8 * 16);
    quant_f32_kernel<<<1, 256, 0, stream>>>(IN_F(38), WsQ, 12 * 32);
    quant_f32_kernel<<<1, 256, 0, stream>>>(IN_F(40), gWbQ, 64 * 62);
    quant_f32_kernel<<<1, 256, 0, stream>>>(IN_F(42), tWbQ, 64 * 62);
    quant_f32_kernel<<<1, 256, 0, stream>>>(IN_F(54), WfQ, 5 * 5);
    quant_f32_kernel<<<1, 256, 0, stream>>>(IN_F(56), gWaQ, 64 * 44);
    quant_f32_kernel<<<1, 256, 0, stream>>>(IN_F(58), tWaQ, 64 * 44);
    quant_f32_kernel<<<1, 256, 0, stream>>>(IN_F(68), WeQ, 64 * 96);
    quant_f32_kernel<<<1, 256, 0, stream>>>(IN_F(74), WpQ, 8 * 11);
    quant_f32_kernel<<<1, 256, 0, stream>>>(IN_F(76), gWmQ, 32 * 15);
    quant_f32_kernel<<<1, 256, 0, stream>>>(IN_F(78), tWmQ, 32 * 15);

    quant_f16_kernel<<<1, 256, 0, stream>>>(IN_F(82), g1q, 256 * 256);
    quant_f16_kernel<<<1, 256, 0, stream>>>(IN_F(84), g2q, 256 * 256);
    quant_f16_kernel<<<1, 256, 0, stream>>>(IN_F(86), t1q, 256 * 256);
    quant_f16_kernel<<<1, 256, 0, stream>>>(IN_F(88), t2q, 256 * 256);
    conv_f16_kernel<<<(512 * 256 + 255) / 256, 256, 0, stream>>>(IN_F(90), m1q, 512 * 256);
    conv_f16_kernel<<<(256 * 512 + 255) / 256, 256, 0, stream>>>(IN_F(92), m2q, 256 * 512);

    // ---- encoder ----
    EncP ep;
    ep.low8 = IN_I(0);  ep.high8 = IN_I(1);  ep.align_ = IN_I(2); ep.small_ = IN_I(3);
    ep.delta = IN_I(4); ep.hamm = IN_I(5);
    ep.continuous = IN_F(6); ep.bitfield = IN_F(7); ep.sketch = IN_F(8);
    ep.ph = IN_I(9); ep.off = IN_I(10); ep.cl = IN_I(11); ep.aa = IN_I(12);
    ep.str = IN_I(13); ep.rd = IN_I(14); ep.lc = IN_I(15); ep.ent = IN_I(16);
    ep.aflags = IN_F(17);
    ep.et = IN_I(18); ep.fc = IN_I(19); ep.sc = IN_I(20); ep.of = IN_I(21);
    ep.tt = IN_I(22); ep.rc = IN_I(23);
    ep.pte = IN_F(24); ep.vma = IN_I(25); ep.prot = IN_I(26); ep.zsum = IN_F(27);
    ep.E_low8 = IN_F(28); ep.E_high8 = IN_F(29); ep.E_align = IN_F(30);
    ep.E_small = IN_F(31); ep.E_delta = IN_F(32); ep.E_hamm = IN_F(33);
    ep.bc = IN_F(35); ep.bb = IN_F(37); ep.bs = IN_F(39);
    ep.gb_b = IN_F(41); ep.tb_b = IN_F(43); ep.lng_b = IN_F(44); ep.lnb_b = IN_F(45);
    ep.E_ph = IN_F(46); ep.E_off = IN_F(47); ep.E_cl = IN_F(48); ep.E_aa = IN_F(49);
    ep.E_str = IN_F(50); ep.E_rd = IN_F(51); ep.E_lc = IN_F(52); ep.E_ent = IN_F(53);
    ep.bf = IN_F(55); ep.gb_a = IN_F(57); ep.tb_a = IN_F(59);
    ep.lng_a = IN_F(60); ep.lnb_a = IN_F(61);
    ep.E_et = IN_F(62); ep.E_fc = IN_F(63); ep.E_sc = IN_F(64); ep.E_of = IN_F(65);
    ep.E_tt = IN_F(66); ep.E_rc = IN_F(67); ep.b_e = IN_F(69);
    ep.lng_e = IN_F(70); ep.lnb_e = IN_F(71);
    ep.E_vma = IN_F(72); ep.E_prot = IN_F(73); ep.bp = IN_F(75);
    ep.gb_m = IN_F(77); ep.tb_m = IN_F(79); ep.lng_m = IN_F(80); ep.lnb_m = IN_F(81);
    ep.Wc = WcQ; ep.Wb = WbQ; ep.Ws = WsQ; ep.gW_b = gWbQ; ep.tW_b = tWbQ;
    ep.Wf = WfQ; ep.gW_a = gWaQ; ep.tW_a = tWaQ; ep.W_e = WeQ; ep.Wp = WpQ;
    ep.gW_m = gWmQ; ep.tW_m = tWmQ;
    ep.comb = comb;
    encoder_kernel<<<BROWS / 256, 256, 0, stream>>>(ep);

    // ---- fused WMMA block ----
    GemmP gp;
    gp.comb = comb;
    gp.g1q = g1q; gp.g2q = g2q; gp.t1q = t1q; gp.t2q = t2q; gp.m1q = m1q; gp.m2q = m2q;
    gp.g1b = IN_F(83); gp.g2b = IN_F(85); gp.t1b = IN_F(87); gp.t2b = IN_F(89);
    gp.m1b = IN_F(91); gp.m2b = IN_F(93);
    gp.ln1g = IN_F(94); gp.ln1b = IN_F(95); gp.ln2g = IN_F(96); gp.ln2b = IN_F(97);
    gp.out = (float*)d_out;
    const size_t lds_bytes =
        (32 * 256 + 32 * 512 + 32 * 256 + 32 * 256) * sizeof(_Float16); // 80 KB
    fused_block_kernel<<<BROWS / 32, 256, lds_bytes, stream>>>(gp);

    #undef IN_F
    #undef IN_I
}
